// SequentialModel_63093069578625
// MI455X (gfx1250) — compile-verified, matrix operand order checked by reading, not run on hardware
//
// CDNA5 (gfx1250) persistent DNC-LSTM forward.
// Design: latency-bound recurrent net (2048 dependent GEMMs); weights (~9.7 MB
// f16) live in the 192 MB L2. One persistent kernel, 16 WGs x 256 threads
// (128 wave32 waves), software grid barrier between dependent phases.
// All GEMMs use v_wmma_f32_16x16x32_f16 (f16 in, f32 accumulate), 16x16 tiles.
#include <hip/hip_runtime.h>
#include <cstdint>
#include <cstddef>

#define B_    128
#define T_    64
#define H_    128
#define L_    16
#define G4H   512
#define KPIN  160   // NN_IN=136 padded to 5 k-tiles of 32
#define E_    15
#define NCLS_ 30
#define XI_   40
#define XIP   48    // xi padded to 3 n-tiles of 16
#define FRIK  640
#define CATK  256
#define EPSF  1e-6f
#define CLIPF 20.0f

typedef __attribute__((ext_vector_type(16))) _Float16 v16h;
typedef __attribute__((ext_vector_type(8)))  _Float16 v8h;
typedef __attribute__((ext_vector_type(8)))  float    v8f;

// ---------------- workspace layout (host+device identical) ----------------
struct Layout {
  size_t bar, wih, whh, wif, wout, wfri, wc1, wc2, f16, h16, cst,
         bx0, bx1, gates, xi, cat, h1, mem, link, prec, wws, rws, usg, rvec,
         total;
};
__host__ __device__ constexpr size_t al256(size_t x){ return (x + 255) & ~(size_t)255; }
__host__ __device__ constexpr Layout make_layout() {
  Layout L{}; size_t o = 0;
  L.bar  = o; o += 256;
  L.wih  = o; o = al256(o + (size_t)L_*2*G4H*KPIN*2);
  L.whh  = o; o = al256(o + (size_t)L_*2*G4H*H_*2);
  L.wif  = o; o = al256(o + (size_t)XIP*H_*2);
  L.wout = o; o = al256(o + (size_t)H_*KPIN*2);
  L.wfri = o; o = al256(o + (size_t)H_*FRIK*2);
  L.wc1  = o; o = al256(o + (size_t)H_*CATK*2);
  L.wc2  = o; o = al256(o + (size_t)32*H_*2);
  L.f16  = o; o = al256(o + (size_t)B_*FRIK*2);
  L.h16  = o; o = al256(o + (size_t)L_*2*B_*H_*2);
  L.cst  = o; o = al256(o + (size_t)L_*2*B_*H_*4);
  L.bx0  = o; o = al256(o + (size_t)B_*KPIN*2);
  L.bx1  = o; o = al256(o + (size_t)B_*KPIN*2);
  L.gates= o; o = al256(o + (size_t)B_*G4H*4);
  L.xi   = o; o = al256(o + (size_t)B_*XIP*4);
  L.cat  = o; o = al256(o + (size_t)B_*CATK*2);
  L.h1   = o; o = al256(o + (size_t)B_*H_*2);
  L.mem  = o; o = al256(o + (size_t)B_*64*4);
  L.link = o; o = al256(o + (size_t)B_*64*4);
  L.prec = o; o = al256(o + (size_t)B_*8*4);
  L.wws  = o; o = al256(o + (size_t)B_*8*4);
  L.rws  = o; o = al256(o + (size_t)B_*8*4);
  L.usg  = o; o = al256(o + (size_t)B_*8*4);
  L.rvec = o; o = al256(o + (size_t)B_*8*4);
  L.total = o;
  return L;
}

// ---------------- device helpers ----------------
__device__ __forceinline__ float sigf(float x){ return 1.f / (1.f + __expf(-x)); }
__device__ __forceinline__ float softplusf_(float x){ return (x > 30.f) ? x : log1pf(__expf(x)); }

// A/B fragment for v_wmma_f32_16x16x32_f16 (wave32):
// lane = half*16 + r ; element i<8 -> K = half*8 + i ; i>=8 -> K = 16 + half*8 + (i-8)
__device__ __forceinline__ v16h load_frag(const _Float16* base) {
  v8h lo = *(const v8h*)(base);
  v8h hi = *(const v8h*)(base + 16);
  return __builtin_shufflevector(lo, hi, 0,1,2,3,4,5,6,7,8,9,10,11,12,13,14,15);
}

// D = A(16xK) * Wt(Kx16) + acc, with A[m][k] row-major (lda halves) and the
// weight stored natively as W[n][k] row-major (ldw halves): B = W^T.
__device__ __forceinline__ v8f wmma_rowmajor(const _Float16* A, int lda,
                                             const _Float16* W, int ldw,
                                             int ktiles, v8f acc, int lane) {
  const int r = lane & 15, half = lane >> 4;
  const _Float16* ap = A + (size_t)r * lda + half * 8;
  const _Float16* wp = W + (size_t)r * ldw + half * 8;
  #pragma unroll
  for (int kt = 0; kt < ktiles; ++kt) {
    v16h a = load_frag(ap + kt * 32);
    v16h b = load_frag(wp + kt * 32);
    acc = __builtin_amdgcn_wmma_f32_16x16x32_f16(false, a, false, b,
                                                 (short)0, acc, false, false);
  }
  return acc;
}

// C/D layout: lane=half*16+r, VGPR v -> M = v + 8*half, N = r
__device__ __forceinline__ void store_tile_f32(float* dst, int ldd, v8f acc, int lane) {
  const int r = lane & 15, half = lane >> 4;
  #pragma unroll
  for (int v = 0; v < 8; ++v) dst[(size_t)(v + 8*half) * ldd + r] = acc[v];
}
__device__ __forceinline__ void store_tile_f16(_Float16* dst, int ldd, v8f acc,
                                               int lane, bool relu) {
  const int r = lane & 15, half = lane >> 4;
  #pragma unroll
  for (int v = 0; v < 8; ++v) {
    float x = acc[v];
    if (relu) x = fmaxf(x, 0.f);
    dst[(size_t)(v + 8*half) * ldd + r] = (_Float16)x;
  }
}

// software grid barrier (generation counter, device-scope fences)
__device__ __forceinline__ void gsync(unsigned* cnt, unsigned* gen) {
  __threadfence();                 // every thread: complete own stores (device scope)
  __syncthreads();
  if (threadIdx.x == 0) {
    unsigned g = __atomic_load_n(gen, __ATOMIC_RELAXED);
    if (atomicAdd(cnt, 1u) == gridDim.x - 1) {
      __atomic_store_n(cnt, 0u, __ATOMIC_RELAXED);
      __atomic_store_n(gen, g + 1u, __ATOMIC_RELEASE);
    } else {
      while (__atomic_load_n(gen, __ATOMIC_ACQUIRE) == g)
        __builtin_amdgcn_s_sleep(1);
    }
  }
  __syncthreads();
  __threadfence();                 // acquire: invalidate stale near caches
}

// gate GEMM: gates = xin @ Wih^T + hprev @ Whh^T + (bih+bhh); 8x32 tiles, 2/wave
__device__ __forceinline__ void gate_gemm(const _Float16* xin, int kx,
                                          const _Float16* hprev,
                                          const _Float16* wih, const _Float16* whh,
                                          const float* bih, const float* bhh,
                                          float* gates, int gwave, int lane) {
  const int r = lane & 15;
  #pragma unroll
  for (int rep = 0; rep < 2; ++rep) {
    const int tile = gwave + rep * 128;       // 0..255
    const int mt = tile >> 5, nt = tile & 31; // 8 x 32 tiles
    const float bias = bih[nt*16 + r] + bhh[nt*16 + r];
    v8f acc;
    #pragma unroll
    for (int v = 0; v < 8; ++v) acc[v] = bias;
    acc = wmma_rowmajor(xin + (size_t)mt*16*KPIN, KPIN,
                        wih + (size_t)nt*16*KPIN, KPIN, kx, acc, lane);
    acc = wmma_rowmajor(hprev + (size_t)mt*16*H_, H_,
                        whh + (size_t)nt*16*H_, H_, 4, acc, lane);
    store_tile_f32(gates + (size_t)mt*16*G4H + nt*16, G4H, acc, lane);
  }
}

__device__ __forceinline__ void cell_elem(const float* gates, float* c,
                                          _Float16* hdst, _Float16* xdst,
                                          bool clip, int gtid, int nthr) {
  for (int i = gtid; i < B_*H_; i += nthr) {
    const int b = i >> 7, j = i & 127;
    const float* g = gates + (size_t)b * G4H;
    const float gi = g[j], gf = g[128+j], gg = g[256+j], go = g[384+j];
    const float cp = c[i];
    const float c2 = sigf(gf)*cp + sigf(gi)*tanhf(gg);
    const float h2 = sigf(go)*tanhf(c2);
    c[i] = c2;
    hdst[i] = (_Float16)h2;
    float o = h2;
    if (clip) o = fminf(fmaxf(o, -CLIPF), CLIPF);
    xdst[(size_t)b*KPIN + j] = (_Float16)o;   // cols 128..159 stay zero
  }
}

__device__ __forceinline__ void emb_fill(const int* x, const float* embed,
                                         _Float16* bx0, int t, int gtid, int nthr) {
  for (int i = gtid; i < B_*KPIN; i += nthr) {
    const int b = i / KPIN, col = i - b*KPIN;
    float v = 0.f;
    if (col < E_) v = embed[x[b*T_ + t]*E_ + col];
    bx0[i] = (_Float16)v;
  }
}

// cosine-similarity content addressing + softmax over 8 cells
__device__ __forceinline__ void content8(const float m[8][8], const float k[8],
                                         float strength, float outw[8]) {
  float kn = 0.f;
  #pragma unroll
  for (int c = 0; c < 8; ++c) kn += k[c]*k[c];
  const float kinv = 1.f / (sqrtf(kn) + EPSF);
  float sim[8]; float mx = -1e30f;
  #pragma unroll
  for (int r = 0; r < 8; ++r) {
    float n2 = 0.f, d = 0.f;
    #pragma unroll
    for (int c = 0; c < 8; ++c) { n2 += m[r][c]*m[r][c]; d += k[c]*m[r][c]; }
    sim[r] = d * kinv / (sqrtf(n2) + EPSF) * strength;
    mx = fmaxf(mx, sim[r]);
  }
  float s = 0.f;
  #pragma unroll
  for (int r = 0; r < 8; ++r) { float e = __expf(sim[r]-mx); outw[r] = e; s += e; }
  const float inv = 1.f / s;
  #pragma unroll
  for (int r = 0; r < 8; ++r) outw[r] *= inv;
}

__device__ void mem_step(const float* xv, float* M, float* Lk, float* Pr,
                         float* Ww, float* Rw, float* Us, float* rvec_out) {
  float rk[8], wk[8], er[8], wv[8];
  #pragma unroll
  for (int j = 0; j < 8; ++j) {
    rk[j] = tanhf(xv[j]);      wk[j] = tanhf(xv[9+j]);
    er[j] = sigf(xv[18+j]);    wv[j] = tanhf(xv[26+j]);
  }
  const float rs  = softplusf_(xv[8]);
  const float wsS = softplusf_(xv[17]);
  const float fg = sigf(xv[34]), ag = sigf(xv[35]), wg = sigf(xv[36]);
  const float m0 = xv[37], m1 = xv[38], m2 = xv[39];
  const float mm = fmaxf(m0, fmaxf(m1, m2));
  const float e0 = __expf(m0-mm), e1 = __expf(m1-mm), e2 = __expf(m2-mm);
  const float es = 1.f / (e0 + e1 + e2);
  const float mode_b = e0*es, mode_f = e1*es, mode_c = e2*es;

  float mem[8][8], lk[8][8], pr[8], rw[8], ww[8], us[8];
  #pragma unroll
  for (int i = 0; i < 8; ++i) {
    pr[i]=Pr[i]; rw[i]=Rw[i]; ww[i]=Ww[i]; us[i]=Us[i];
    #pragma unroll
    for (int j = 0; j < 8; ++j) { mem[i][j]=M[i*8+j]; lk[i][j]=Lk[i*8+j]; }
  }
  // usage update (R=1, single write head)
  #pragma unroll
  for (int j = 0; j < 8; ++j) {
    us[j] = us[j] + (1.f - us[j]) * ww[j];
    us[j] *= (1.f - fg * rw[j]);
  }
  float wcw[8]; content8(mem, wk, wsS, wcw);
  // allocation weighting: stable ascending sort of u (adjacent-swap, static idx)
  float u[8]; int idx[8];
  #pragma unroll
  for (int j = 0; j < 8; ++j) { u[j] = EPSF + (1.f - EPSF)*us[j]; idx[j] = j; }
  #pragma unroll
  for (int i = 1; i < 8; ++i)
    #pragma unroll
    for (int j = 7; j >= 1; --j)
      if (u[j-1] > u[j]) {
        float tu = u[j-1]; u[j-1] = u[j]; u[j] = tu;
        int   ti = idx[j-1]; idx[j-1] = idx[j]; idx[j] = ti;
      }
  float alloc[8]; float run = 1.f;
  #pragma unroll
  for (int k = 0; k < 8; ++k) {
    const float a = (1.f - u[k]) * run; run *= u[k];
    #pragma unroll
    for (int j = 0; j < 8; ++j) if (idx[k] == j) alloc[j] = a;
  }
  #pragma unroll
  for (int j = 0; j < 8; ++j) ww[j] = wg * (ag*alloc[j] + (1.f-ag)*wcw[j]);
  // memory write
  #pragma unroll
  for (int m = 0; m < 8; ++m)
    #pragma unroll
    for (int c = 0; c < 8; ++c)
      mem[m][c] = mem[m][c]*(1.f - ww[m]*er[c]) + ww[m]*wv[c];
  // temporal link
  #pragma unroll
  for (int i = 0; i < 8; ++i)
    #pragma unroll
    for (int j = 0; j < 8; ++j)
      lk[i][j] = (i == j) ? 0.f : ((1.f - ww[i] - ww[j])*lk[i][j] + ww[i]*pr[j]);
  float sw = 0.f;
  #pragma unroll
  for (int j = 0; j < 8; ++j) sw += ww[j];
  #pragma unroll
  for (int j = 0; j < 8; ++j) pr[j] = (1.f - sw)*pr[j] + ww[j];
  // read weighting (content on updated mem; fwd/bwd with OLD rw)
  float cw[8]; content8(mem, rk, rs, cw);
  float fw[8], bw[8];
  #pragma unroll
  for (int i = 0; i < 8; ++i) {
    float s1 = 0.f, s2 = 0.f;
    #pragma unroll
    for (int j = 0; j < 8; ++j) { s1 += lk[i][j]*rw[j]; s2 += rw[j]*lk[j][i]; }
    fw[i] = s1; bw[i] = s2;
  }
  float rwn[8];
  #pragma unroll
  for (int j = 0; j < 8; ++j) rwn[j] = mode_c*cw[j] + mode_f*fw[j] + mode_b*bw[j];
  float rv[8];
  #pragma unroll
  for (int c = 0; c < 8; ++c) {
    float s = 0.f;
    #pragma unroll
    for (int m2 = 0; m2 < 8; ++m2) s += rwn[m2]*mem[m2][c];
    rv[c] = s;
  }
  #pragma unroll
  for (int i = 0; i < 8; ++i) {
    Pr[i]=pr[i]; Rw[i]=rwn[i]; Ww[i]=ww[i]; Us[i]=us[i]; rvec_out[i]=rv[i];
    #pragma unroll
    for (int j = 0; j < 8; ++j) { M[i*8+j]=mem[i][j]; Lk[i*8+j]=lk[i][j]; }
  }
}

// ---------------- kernels ----------------
__global__ void pack_f16_k(const float* __restrict__ src, _Float16* __restrict__ dst,
                           int rows, int ks, int kd) {
  const int i = blockIdx.x * blockDim.x + threadIdx.x;
  if (i >= rows * ks) return;
  const int r = i / ks, k = i - r * ks;
  dst[(size_t)r * kd + k] = (_Float16)src[i];
}

struct ModelArgs {
  unsigned char* ws;
  const int* x; const float* embed;
  const float* b_ih; const float* b_hh; const float* b_if;
  const float* b_out; const float* b_fri; const float* b_c1; const float* b_c2;
  float* out;
};

__global__ __launch_bounds__(256) void dnc_persistent_k(ModelArgs A) {
  constexpr Layout Lw = make_layout();
  unsigned char* ws = A.ws;
  _Float16* wih16 = (_Float16*)(ws + Lw.wih);
  _Float16* whh16 = (_Float16*)(ws + Lw.whh);
  _Float16* wif16 = (_Float16*)(ws + Lw.wif);
  _Float16* wout16= (_Float16*)(ws + Lw.wout);
  _Float16* wfri16= (_Float16*)(ws + Lw.wfri);
  _Float16* wc116 = (_Float16*)(ws + Lw.wc1);
  _Float16* wc216 = (_Float16*)(ws + Lw.wc2);
  _Float16* f16f  = (_Float16*)(ws + Lw.f16);
  _Float16* h16   = (_Float16*)(ws + Lw.h16);
  float*    cst   = (float*)   (ws + Lw.cst);
  _Float16* bx0   = (_Float16*)(ws + Lw.bx0);
  _Float16* bx1   = (_Float16*)(ws + Lw.bx1);
  float*    gates = (float*)   (ws + Lw.gates);
  float*    xi    = (float*)   (ws + Lw.xi);
  _Float16* cat16 = (_Float16*)(ws + Lw.cat);
  _Float16* h116  = (_Float16*)(ws + Lw.h1);
  float*    memst = (float*)   (ws + Lw.mem);
  float*    lnkst = (float*)   (ws + Lw.link);
  float*    prec  = (float*)   (ws + Lw.prec);
  float*    wws   = (float*)   (ws + Lw.wws);
  float*    rws   = (float*)   (ws + Lw.rws);
  float*    usg   = (float*)   (ws + Lw.usg);
  float*    rvec  = (float*)   (ws + Lw.rvec);
  unsigned* cnt   = (unsigned*)(ws + Lw.bar);
  unsigned* gen   = cnt + 1;

  const int tid   = threadIdx.x;
  const int lane  = tid & 31;
  const int gwave = blockIdx.x * 8 + (tid >> 5);   // 0..127
  const int gtid  = blockIdx.x * 256 + tid;
  const int nthr  = gridDim.x * 256;

  emb_fill(A.x, A.embed, bx0, 0, gtid, nthr);
  gsync(cnt, gen);

  for (int t = 0; t < T_; ++t) {
    for (int l = 0; l < L_; ++l) {
      const _Float16* wih0 = wih16 + (size_t)(2*l)   * G4H * KPIN;
      const _Float16* wih1 = wih16 + (size_t)(2*l+1) * G4H * KPIN;
      const _Float16* whh0 = whh16 + (size_t)(2*l)   * G4H * H_;
      const _Float16* whh1 = whh16 + (size_t)(2*l+1) * G4H * H_;
      const float* bi0 = A.b_ih + (2*l)*G4H;   const float* bh0 = A.b_hh + (2*l)*G4H;
      const float* bi1 = A.b_ih + (2*l+1)*G4H; const float* bh1 = A.b_hh + (2*l+1)*G4H;
      _Float16* h0 = h16 + (size_t)(2*l)   * B_ * H_;
      _Float16* h1 = h16 + (size_t)(2*l+1) * B_ * H_;
      float* c0 = cst + (size_t)(2*l)   * B_ * H_;
      float* c1 = cst + (size_t)(2*l+1) * B_ * H_;

      // P1: cell0 gates (kx=5: 136-pad input)
      gate_gemm(bx0, 5, h0, wih0, whh0, bi0, bh0, gates, gwave, lane);
      gsync(cnt, gen);
      // P2: cell0 pointwise -> c0, h0, bx1 (h0 padded)
      cell_elem(gates, c0, h0, bx1, false, gtid, nthr);
      gsync(cnt, gen);
      // P3: cell1 gates (kx=4: only 128 live cols of bx1)
      gate_gemm(bx1, 4, h1, wih1, whh1, bi1, bh1, gates, gwave, lane);
      gsync(cnt, gen);
      // P4: cell1 pointwise -> c1, h1, bx0[:, :128] = clip(h1)
      cell_elem(gates, c1, h1, bx0, true, gtid, nthr);
      gsync(cnt, gen);
      // P5: xi = out @ W_if^T + b_if  (8x3 tiles)
      if (gwave < 24) {
        const int mt = gwave / 3, nt = gwave % 3;
        const int r = lane & 15, n = nt*16 + r;
        const float bias = (n < XI_) ? A.b_if[n] : 0.f;
        v8f acc;
        #pragma unroll
        for (int v = 0; v < 8; ++v) acc[v] = bias;
        acc = wmma_rowmajor(bx0 + (size_t)mt*16*KPIN, KPIN,
                            wif16 + (size_t)nt*16*H_, H_, 4, acc, lane);
        store_tile_f32(xi + (size_t)mt*16*XIP + nt*16, XIP, acc, lane);
      }
      gsync(cnt, gen);
      // P6: memory step (1 thread / batch row) + next-step embedding fill
      if (gtid < B_) {
        mem_step(xi + (size_t)gtid*XIP, memst + gtid*64, lnkst + gtid*64,
                 prec + gtid*8, wws + gtid*8, rws + gtid*8, usg + gtid*8,
                 rvec + gtid*8);
        if (l < L_-1 || t == T_-1) {
          #pragma unroll
          for (int c = 0; c < 8; ++c)
            bx0[(size_t)gtid*KPIN + H_ + c] = (_Float16)rvec[gtid*8 + c];
        }
      }
      if (l == L_-1 && t < T_-1) emb_fill(A.x, A.embed, bx0, t+1, gtid, nthr);
      gsync(cnt, gen);
    }
  }

  // ---- classifier head (bx0 now holds inp of the last layer/timestep) ----
  // F1: z = inp @ W_out^T + b_out -> cat[:, :128];  f_out = relu(f @ W_fri^T + b_fri) -> cat[:, 128:]
  if (gwave < 64) {
    const int mt = gwave >> 3, nt = gwave & 7;
    const int r = lane & 15;
    const float bias = A.b_out[nt*16 + r];
    v8f acc;
    #pragma unroll
    for (int v = 0; v < 8; ++v) acc[v] = bias;
    acc = wmma_rowmajor(bx0 + (size_t)mt*16*KPIN, KPIN,
                        wout16 + (size_t)nt*16*KPIN, KPIN, 5, acc, lane);
    store_tile_f16(cat16 + (size_t)mt*16*CATK + nt*16, CATK, acc, lane, false);
  } else {
    const int w2 = gwave - 64;
    const int mt = w2 >> 3, nt = w2 & 7;
    const int r = lane & 15;
    const float bias = A.b_fri[nt*16 + r];
    v8f acc;
    #pragma unroll
    for (int v = 0; v < 8; ++v) acc[v] = bias;
    acc = wmma_rowmajor(f16f + (size_t)mt*16*FRIK, FRIK,
                        wfri16 + (size_t)nt*16*FRIK, FRIK, 20, acc, lane);
    store_tile_f16(cat16 + (size_t)mt*16*CATK + H_ + nt*16, CATK, acc, lane, true);
  }
  gsync(cnt, gen);
  // F2: h1 = relu(cat @ W_c1^T + b_c1)
  if (gwave < 64) {
    const int mt = gwave >> 3, nt = gwave & 7;
    const int r = lane & 15;
    const float bias = A.b_c1[nt*16 + r];
    v8f acc;
    #pragma unroll
    for (int v = 0; v < 8; ++v) acc[v] = bias;
    acc = wmma_rowmajor(cat16 + (size_t)mt*16*CATK, CATK,
                        wc116 + (size_t)nt*16*CATK, CATK, 8, acc, lane);
    store_tile_f16(h116 + (size_t)mt*16*H_ + nt*16, H_, acc, lane, true);
  }
  gsync(cnt, gen);
  // F3: out = h1 @ W_c2^T + b_c2  (N padded 30->32; write n<30)
  if (gwave < 16) {
    const int mt = gwave >> 1, nt = gwave & 1;
    const int r = lane & 15, half = lane >> 4;
    const int n = nt*16 + r;
    const float bias = (n < NCLS_) ? A.b_c2[n] : 0.f;
    v8f acc;
    #pragma unroll
    for (int v = 0; v < 8; ++v) acc[v] = bias;
    acc = wmma_rowmajor(h116 + (size_t)mt*16*H_, H_,
                        wc216 + (size_t)nt*16*H_, H_, 4, acc, lane);
    if (n < NCLS_) {
      #pragma unroll
      for (int v = 0; v < 8; ++v)
        A.out[(size_t)(mt*16 + v + 8*half)*NCLS_ + n] = acc[v];
    }
  }
}

// ---------------- host launcher ----------------
extern "C" void kernel_launch(void* const* d_in, const int* in_sizes, int n_in,
                              void* d_out, int out_size, void* d_ws, size_t ws_size,
                              hipStream_t stream) {
  (void)in_sizes; (void)n_in; (void)out_size; (void)ws_size;
  constexpr Layout Lw = make_layout();
  unsigned char* ws = (unsigned char*)d_ws;

  // zero workspace: weight padding, all recurrent/memory state, barrier
  hipMemsetAsync(ws, 0, Lw.total, stream);

  auto pack = [&](const void* src, size_t dstoff, int rows, int ks, int kd) {
    const int n = rows * ks;
    pack_f16_k<<<(n + 255) / 256, 256, 0, stream>>>(
        (const float*)src, (_Float16*)(ws + dstoff), rows, ks, kd);
  };
  pack(d_in[3],  Lw.wih,  L_*2*G4H, 136, KPIN);  // W_ih
  pack(d_in[4],  Lw.whh,  L_*2*G4H, 128, 128);   // W_hh
  pack(d_in[7],  Lw.wif,  XI_,      128, 128);   // W_if  (rows 40..47 zero)
  pack(d_in[9],  Lw.wout, H_,       136, KPIN);  // W_out
  pack(d_in[11], Lw.wfri, H_,       640, 640);   // W_fri
  pack(d_in[13], Lw.wc1,  H_,       256, 256);   // W_c1
  pack(d_in[15], Lw.wc2,  NCLS_,    128, 128);   // W_c2  (rows 30,31 zero)
  pack(d_in[1],  Lw.f16,  B_,       640, 640);   // f -> f16

  ModelArgs A;
  A.ws    = ws;
  A.x     = (const int*)d_in[0];
  A.embed = (const float*)d_in[2];
  A.b_ih  = (const float*)d_in[5];
  A.b_hh  = (const float*)d_in[6];
  A.b_if  = (const float*)d_in[8];
  A.b_out = (const float*)d_in[10];
  A.b_fri = (const float*)d_in[12];
  A.b_c1  = (const float*)d_in[14];
  A.b_c2  = (const float*)d_in[16];
  A.out   = (float*)d_out;

  dnc_persistent_k<<<16, 256, 0, stream>>>(A);
}
